// head_attention_v3_35845797052903
// MI455X (gfx1250) — compile-verified, matrix operand order checked by reading, not run on hardware
//
#include <hip/hip_runtime.h>
#include <hip/hip_bf16.h>

typedef __attribute__((ext_vector_type(16))) _Float16 v16h;
typedef __attribute__((ext_vector_type(8)))  float    v8f;
typedef __attribute__((ext_vector_type(4)))  float    f32x4;

#define DEV __device__ __forceinline__

DEV v8f wmma16(v16h a, v16h b, v8f c) {
  // D = A(16x32 f16) x B(32x16 f16) + C(16x16 f32)
  return __builtin_amdgcn_wmma_f32_16x16x32_f16(false, a, false, b, (short)0, c,
                                                false, false);
}

// Load a 16x32 WMMA fragment (A layout; B uses the same per-lane mapping with
// the 16-index being N instead of M). Element (m, k) lives at p[m*rs + k*ks].
// Per ISA 7.12.2: lane L -> m = L%16, K-halves are pairs (k, k+1) with
// k = {0,2,4,6,16,18,20,22} + (L/16)*8.
template <typename T>
DEV v16h load_frag(const T* p, int rs, int ks, int lane) {
  const int m = lane & 15, g = lane >> 4;
  v16h x;
#pragma unroll
  for (int v = 0; v < 8; ++v) {
    const int k = ((v & 4) ? (16 + 2 * (v & 3)) : (2 * v)) + g * 8;
    x[2 * v]     = (_Float16)p[m * rs + k * ks];
    x[2 * v + 1] = (_Float16)p[m * rs + (k + 1) * ks];
  }
  return x;
}

DEV void ld16(const float* p, float* r) {
#pragma unroll
  for (int q = 0; q < 4; ++q) {
    f32x4 v = ((const f32x4*)p)[q];
    r[4 * q + 0] = v.x; r[4 * q + 1] = v.y;
    r[4 * q + 2] = v.z; r[4 * q + 3] = v.w;
  }
}
// Non-temporal (streaming) variants: e is read once, e_new written once.
// TH=NT keeps the reused CnT/CqT/Vn tables resident in the 192MB L2 while the
// 268MB of stream traffic passes through.
DEV void ld16_nt(const float* p, float* r) {
#pragma unroll
  for (int q = 0; q < 4; ++q) {
    f32x4 v = __builtin_nontemporal_load(((const f32x4*)p) + q);
    r[4 * q + 0] = v.x; r[4 * q + 1] = v.y;
    r[4 * q + 2] = v.z; r[4 * q + 3] = v.w;
  }
}
DEV void st16_nt(float* p, const float* r) {
#pragma unroll
  for (int q = 0; q < 4; ++q) {
    f32x4 v = {r[4 * q + 0], r[4 * q + 1], r[4 * q + 2], r[4 * q + 3]};
    __builtin_nontemporal_store(v, ((f32x4*)p) + q);
  }
}

// ---------------------------------------------------------------------------
// K1: Q/K/V projections of x. Q,K stored f16 row-major (B,512,64); V stored
// f16 transposed Vt (B,64,512) so P@V B-fragments are contiguous in K.
// Unified strided store: dst[row*rs + col*cs], no divergent branches.
// ---------------------------------------------------------------------------
__global__ void k_proj_qkv(const float* __restrict__ x,
                           const float* __restrict__ Wq, const float* __restrict__ bq,
                           const float* __restrict__ Wk, const float* __restrict__ bk,
                           const float* __restrict__ Wv, const float* __restrict__ bv,
                           _Float16* __restrict__ Qh, _Float16* __restrict__ Kh,
                           _Float16* __restrict__ Vt) {
  const int lane = threadIdx.x & 31;
  const int task = blockIdx.x * (blockDim.x >> 5) + (threadIdx.x >> 5); // 0..191
  const int proj = task % 3;
  const int rtg  = task / 3;          // 0..63
  const int b    = rtg >> 5;
  const int row0 = (rtg & 31) * 16;
  const int m = lane & 15, g = lane >> 4;
  const float* W    = proj == 0 ? Wq : proj == 1 ? Wk : Wv;
  const float* bias = proj == 0 ? bq : proj == 1 ? bk : bv;
  _Float16* dst = proj == 0 ? (Qh + b * 512 * 64)
                : proj == 1 ? (Kh + b * 512 * 64)
                            : (Vt + b * 64 * 512);
  const int drs = (proj == 2) ? 1 : 64;    // stride per output row
  const int dcs = (proj == 2) ? 512 : 1;   // stride per output col

  const float* ab = x + (b * 512 + row0) * 64;
  v16h a0 = load_frag(ab,      64, 1, lane);
  v16h a1 = load_frag(ab + 32, 64, 1, lane);
#pragma unroll
  for (int ct = 0; ct < 4; ++ct) {
    // Bt[n][k] = W[k][n] (weights are (din,dout) row-major) -> rs=1, ks=64
    v16h b0 = load_frag(W + ct * 16,            1, 64, lane);
    v16h b1 = load_frag(W + ct * 16 + 32 * 64,  1, 64, lane);
    v8f acc = {};
    acc = wmma16(a0, b0, acc);
    acc = wmma16(a1, b1, acc);
    const float bb = bias[ct * 16 + m];
    const int col = ct * 16 + m;
#pragma unroll
    for (int r = 0; r < 8; ++r) {
      const int row = row0 + r + 8 * g;
      dst[row * drs + col * dcs] = (_Float16)(acc[r] + bb);
    }
  }
}

// ---------------------------------------------------------------------------
// K2: self-attention x_hat = softmax(QK^T/8) V.  One wave per 16-query tile.
// ---------------------------------------------------------------------------
__global__ void __launch_bounds__(32)
k_attn(const _Float16* __restrict__ Qh, const _Float16* __restrict__ Kh,
       const _Float16* __restrict__ Vt, _Float16* __restrict__ Xh) {
  __shared__ float    sS[16 * 512];
  __shared__ _Float16 sP[16 * 512];
  __shared__ float    sL[16];
  const int lane = threadIdx.x & 31;
  const int b  = blockIdx.x >> 5;
  const int q0 = (blockIdx.x & 31) * 16;
  const int m = lane & 15, g = lane >> 4;

  const _Float16* qb = Qh + (b * 512 + q0) * 64;
  v16h a0 = load_frag(qb,      64, 1, lane);
  v16h a1 = load_frag(qb + 32, 64, 1, lane);

  for (int kt = 0; kt < 32; ++kt) {
    const _Float16* kb = Kh + (b * 512 + kt * 16) * 64;  // Bt[n=key][k=feat]
    v8f acc = {};
    acc = wmma16(a0, load_frag(kb,      64, 1, lane), acc);
    acc = wmma16(a1, load_frag(kb + 32, 64, 1, lane), acc);
#pragma unroll
    for (int r = 0; r < 8; ++r)
      sS[(r + 8 * g) * 512 + kt * 16 + m] = acc[r] * 0.125f;
  }
  __syncthreads();

  {  // softmax: 2 lanes per row, 256 cols each
    const int row = lane & 15;
    const int c0  = (lane >> 4) * 256;
    float mx = -3.0e38f;
    for (int c = 0; c < 256; ++c) mx = fmaxf(mx, sS[row * 512 + c0 + c]);
    mx = fmaxf(mx, __shfl_xor(mx, 16));
    float l = 0.f;
    for (int c = 0; c < 256; ++c) {
      const float ev = __expf(sS[row * 512 + c0 + c] - mx);
      sP[row * 512 + c0 + c] = (_Float16)ev;
      l += ev;
    }
    l += __shfl_xor(l, 16);
    if ((lane >> 4) == 0) sL[row] = l;
  }
  __syncthreads();

#pragma unroll
  for (int ct = 0; ct < 4; ++ct) {
    v8f acc = {};
    for (int kc = 0; kc < 16; ++kc) {
      v16h a  = load_frag(sP + kc * 32, 512, 1, lane);                      // P 16x32
      v16h bb = load_frag(Vt + (b * 64 + ct * 16) * 512 + kc * 32, 512, 1, lane);
      acc = wmma16(a, bb, acc);
    }
#pragma unroll
    for (int r = 0; r < 8; ++r) {
      const int row = r + 8 * g;
      Xh[(b * 512 + q0 + row) * 64 + ct * 16 + m] =
          (_Float16)(acc[r] / sL[row]);
    }
  }
}

// ---------------------------------------------------------------------------
// K3: Kn / Qn / Vn projections of x_hat (f16 in; Kn,Qn f16, Vn f32 out).
// ---------------------------------------------------------------------------
__global__ void k_proj_node(const _Float16* __restrict__ Xh,
                            const float* __restrict__ Wkn, const float* __restrict__ bkn,
                            const float* __restrict__ Wqn, const float* __restrict__ bqn,
                            const float* __restrict__ Wvn, const float* __restrict__ bvn,
                            _Float16* __restrict__ Kn, _Float16* __restrict__ Qn,
                            float* __restrict__ Vn) {
  const int lane = threadIdx.x & 31;
  const int task = blockIdx.x * (blockDim.x >> 5) + (threadIdx.x >> 5); // 0..191
  const int proj = task % 3;
  const int rtg  = task / 3;
  const int b    = rtg >> 5;
  const int row0 = (rtg & 31) * 16;
  const int m = lane & 15, g = lane >> 4;
  const float* W    = proj == 0 ? Wkn : proj == 1 ? Wqn : Wvn;
  const float* bias = proj == 0 ? bkn : proj == 1 ? bqn : bvn;
  _Float16* dsth = (proj == 0 ? Kn : Qn);

  const _Float16* ab = Xh + (b * 512 + row0) * 64;
  v16h a0 = load_frag(ab,      64, 1, lane);
  v16h a1 = load_frag(ab + 32, 64, 1, lane);
#pragma unroll
  for (int ct = 0; ct < 4; ++ct) {
    v16h b0 = load_frag(W + ct * 16,           1, 64, lane);
    v16h b1 = load_frag(W + ct * 16 + 32 * 64, 1, 64, lane);
    v8f acc = {};
    acc = wmma16(a0, b0, acc);
    acc = wmma16(a1, b1, acc);
    const float bb = bias[ct * 16 + m];
    if (proj < 2) {
#pragma unroll
      for (int r = 0; r < 8; ++r)
        dsth[(b * 512 + row0 + r + 8 * g) * 64 + ct * 16 + m] =
            (_Float16)(acc[r] + bb);
    } else {
#pragma unroll
      for (int r = 0; r < 8; ++r)
        Vn[(b * 512 + row0 + r + 8 * g) * 64 + ct * 16 + m] = acc[r] + bb;
    }
  }
}

// ---------------------------------------------------------------------------
// K4: CnT = Kn @ Wqe^T ; CqT = Qn @ Wke^T   (folds the edge projections).
// CnT[b,j,d] = sum_o Kn[b,j,o] * Wqe[d,o]  -> Bt[n=d][k=o] = W[d*64+o].
// ---------------------------------------------------------------------------
__global__ void k_cmat(const _Float16* __restrict__ Kn, const _Float16* __restrict__ Qn,
                       const float* __restrict__ Wqe, const float* __restrict__ Wke,
                       float* __restrict__ CnT, float* __restrict__ CqT) {
  const int lane = threadIdx.x & 31;
  const int task = blockIdx.x * (blockDim.x >> 5) + (threadIdx.x >> 5); // 0..127
  const int mat = task & 1;
  const int rtg = task >> 1;
  const int b    = rtg >> 5;
  const int row0 = (rtg & 31) * 16;
  const int m = lane & 15, g = lane >> 4;
  const _Float16* A = (mat == 0 ? Kn : Qn) + (b * 512 + row0) * 64;
  const float* W = mat == 0 ? Wqe : Wke;
  float* out = mat == 0 ? CnT : CqT;

  v16h a0 = load_frag(A,      64, 1, lane);
  v16h a1 = load_frag(A + 32, 64, 1, lane);
#pragma unroll
  for (int ct = 0; ct < 4; ++ct) {
    v16h b0 = load_frag(W + (ct * 16) * 64,      64, 1, lane);
    v16h b1 = load_frag(W + (ct * 16) * 64 + 32, 64, 1, lane);
    v8f acc = {};
    acc = wmma16(a0, b0, acc);
    acc = wmma16(a1, b1, acc);
#pragma unroll
    for (int r = 0; r < 8; ++r)
      out[(b * 512 + row0 + r + 8 * g) * 64 + ct * 16 + m] = acc[r];
  }
}

// K4b: dn[b,j] = bqe . Kn[b,j] ; dq[b,i] = bke . Qn[b,i]
__global__ void k_dvec(const _Float16* __restrict__ Kn, const _Float16* __restrict__ Qn,
                       const float* __restrict__ bqe, const float* __restrict__ bke,
                       float* __restrict__ dn, float* __restrict__ dq) {
  const int id = blockIdx.x * blockDim.x + threadIdx.x; // 0..1023
  if (id < 1024) {
    float s1 = 0.f, s2 = 0.f;
    for (int o = 0; o < 64; ++o) {
      s1 += bqe[o] * (float)Kn[id * 64 + o];
      s2 += bke[o] * (float)Qn[id * 64 + o];
    }
    dn[id] = s1; dq[id] = s2;
  }
}

// ---------------------------------------------------------------------------
// K5: fused edge stage. One block per (b,i): streams e once (NT loads),
// writes e_new once (NT stores), and accumulates the stage-3 softmax
// aggregate online (no e_new re-read). 4 lanes per edge row, 16 floats each.
// ---------------------------------------------------------------------------
__global__ void __launch_bounds__(256)
k_edge(const float* __restrict__ e, const float* __restrict__ Vn,
       const float* __restrict__ CnT, const float* __restrict__ CqT,
       const float* __restrict__ dn, const float* __restrict__ dq,
       float* __restrict__ e_new, float* __restrict__ Agg) {
  __shared__ float vni[64], cni[64], cqi[64];
  __shared__ float rm[256], rl[256];
  __shared__ float ragg[256 * 16];
  const int t = threadIdx.x;
  const int b = blockIdx.x >> 9;
  const int i = blockIdx.x & 511;
  const int c = t & 3;        // column slice (16 floats)
  const int g = t >> 2;       // edge-row group within pass
  const int ibase = (b * 512 + i) * 64;
  if (t < 64) {
    vni[t] = Vn[ibase + t];
    cni[t] = CnT[ibase + t];
    cqi[t] = CqT[ibase + t];
  }
  __syncthreads();
  const float dnbi = dn[b * 512 + i];
  const float dqbi = dq[b * 512 + i];
  const size_t erow0 = (size_t)(b * 512 + i) * 512 * 64;

  // Prefetch first pass's stream slice into cache hierarchy.
  __builtin_prefetch(e + erow0 + g * 64 + c * 16, 0, 0);

  float m = -3.0e38f, l = 0.f, agg[16];
#pragma unroll
  for (int d = 0; d < 16; ++d) agg[d] = 0.f;

  for (int pass = 0; pass < 8; ++pass) {
    const int j = pass * 64 + g;
    const size_t ebase = erow0 + (size_t)j * 64 + c * 16;
    const int    jbase = (b * 512 + j) * 64 + c * 16;
    if (pass < 7)  // prefetch the next pass's slice (global_prefetch_b8)
      __builtin_prefetch(e + ebase + 64 * 64, 0, 0);

    float ev[16], cn[16], vn[16], en[16];
    ld16_nt(e + ebase, ev);       // stream-in, non-temporal
    ld16(CnT + jbase, cn);        // hot tables: normal (cache-resident)
    ld16(Vn + jbase, vn);

    float sip = 0.f, sjp = 0.f;
#pragma unroll
    for (int d = 0; d < 16; ++d) {
      sip += ev[d] * cn[d];
      sjp += ev[d] * cni[c * 16 + d];
    }
    sip += __shfl_xor(sip, 1); sip += __shfl_xor(sip, 2);
    sjp += __shfl_xor(sjp, 1); sjp += __shfl_xor(sjp, 2);
    const float si = (sip + dn[b * 512 + j]) * 0.125f;
    const float sj = (sjp + dnbi) * 0.125f;
    const float ai = 1.f / (1.f + __expf(sj - si));
#pragma unroll
    for (int d = 0; d < 16; ++d)
      en[d] = ai * vn[d] + (1.f - ai) * vni[c * 16 + d];
    st16_nt(e_new + ebase, en);   // stream-out, non-temporal

    float s2p = 0.f;
#pragma unroll
    for (int d = 0; d < 16; ++d) s2p += en[d] * cqi[c * 16 + d];
    s2p += __shfl_xor(s2p, 1); s2p += __shfl_xor(s2p, 2);
    const float s2 = (s2p + dqbi) * 0.125f;

    const float mn = fmaxf(m, s2);
    const float sc = __expf(m - mn);
    const float w  = __expf(s2 - mn);
    l = l * sc + w;
#pragma unroll
    for (int d = 0; d < 16; ++d) agg[d] = agg[d] * sc + w * en[d];
    m = mn;
  }

  rm[t] = m; rl[t] = l;
#pragma unroll
  for (int d = 0; d < 16; ++d) ragg[t * 16 + d] = agg[d];
  __syncthreads();
  for (int off = 128; off >= 4; off >>= 1) {
    if (t < off) {
      const float m2 = rm[t + off], l2 = rl[t + off];
      const float mn = fmaxf(m, m2);
      const float w1 = __expf(m - mn), w2 = __expf(m2 - mn);
      l = l * w1 + l2 * w2;
#pragma unroll
      for (int d = 0; d < 16; ++d)
        agg[d] = agg[d] * w1 + ragg[(t + off) * 16 + d] * w2;
      m = mn;
      rm[t] = m; rl[t] = l;
#pragma unroll
      for (int d = 0; d < 16; ++d) ragg[t * 16 + d] = agg[d];
    }
    __syncthreads();
  }
  if (t < 4) {
    const float inv = 1.f / l;
#pragma unroll
    for (int d = 0; d < 16; ++d) Agg[ibase + t * 16 + d] = agg[d] * inv;
  }
}

// ---------------------------------------------------------------------------
// K6: x_new = Agg @ Wve + bve   (1024 x 64 x 64, WMMA, f32 out)
// ---------------------------------------------------------------------------
__global__ void k_xnew(const float* __restrict__ Agg, const float* __restrict__ Wve,
                       const float* __restrict__ bve, float* __restrict__ xout) {
  const int lane = threadIdx.x & 31;
  const int task = blockIdx.x * (blockDim.x >> 5) + (threadIdx.x >> 5); // 0..63
  const int row0 = task * 16;
  const int m = lane & 15, g = lane >> 4;
  const float* ab = Agg + row0 * 64;
  v16h a0 = load_frag(ab,      64, 1, lane);
  v16h a1 = load_frag(ab + 32, 64, 1, lane);
#pragma unroll
  for (int ct = 0; ct < 4; ++ct) {
    v16h b0 = load_frag(Wve + ct * 16,           1, 64, lane);
    v16h b1 = load_frag(Wve + ct * 16 + 32 * 64, 1, 64, lane);
    v8f acc = {};
    acc = wmma16(a0, b0, acc);
    acc = wmma16(a1, b1, acc);
    const float bb = bve[ct * 16 + m];
#pragma unroll
    for (int r = 0; r < 8; ++r)
      xout[(row0 + r + 8 * g) * 64 + ct * 16 + m] = acc[r] + bb;
  }
}

extern "C" void kernel_launch(void* const* d_in, const int* in_sizes, int n_in,
                              void* d_out, int out_size, void* d_ws, size_t ws_size,
                              hipStream_t stream) {
  (void)in_sizes; (void)n_in; (void)out_size; (void)ws_size;
  const float* x   = (const float*)d_in[0];
  const float* e   = (const float*)d_in[1];
  const float* Wq1 = (const float*)d_in[2];  const float* bq1 = (const float*)d_in[3];
  const float* Wk1 = (const float*)d_in[4];  const float* bk1 = (const float*)d_in[5];
  const float* Wv1 = (const float*)d_in[6];  const float* bv1 = (const float*)d_in[7];
  const float* Wqe = (const float*)d_in[8];  const float* bqe = (const float*)d_in[9];
  const float* Wkn = (const float*)d_in[10]; const float* bkn = (const float*)d_in[11];
  const float* Wvn = (const float*)d_in[12]; const float* bvn = (const float*)d_in[13];
  const float* Wqn = (const float*)d_in[14]; const float* bqn = (const float*)d_in[15];
  const float* Wke = (const float*)d_in[16]; const float* bke = (const float*)d_in[17];
  const float* Wve = (const float*)d_in[18]; const float* bve = (const float*)d_in[19];

  char* ws = (char*)d_ws;
  _Float16* Qh = (_Float16*)(ws + 0);        // 2*512*64 f16
  _Float16* Kh = (_Float16*)(ws + 131072);
  _Float16* Vt = (_Float16*)(ws + 262144);   // transposed (B,64,512)
  _Float16* Xh = (_Float16*)(ws + 393216);
  _Float16* Kn = (_Float16*)(ws + 524288);
  _Float16* Qn = (_Float16*)(ws + 655360);
  float*    Vn  = (float*)(ws + 786432);     // 2*512*64 f32
  float*    CnT = (float*)(ws + 1048576);
  float*    CqT = (float*)(ws + 1310720);
  float*    dnv = (float*)(ws + 1572864);
  float*    dqv = (float*)(ws + 1576960);
  float*    Agg = (float*)(ws + 1581056);

  float* x_new = (float*)d_out;               // (2,512,64)
  float* e_new = (float*)d_out + 65536;       // (2,512,512,64)

  k_proj_qkv<<<24, 256, 0, stream>>>(x, Wq1, bq1, Wk1, bk1, Wv1, bv1, Qh, Kh, Vt);
  k_attn<<<64, 32, 0, stream>>>(Qh, Kh, Vt, Xh);
  k_proj_node<<<24, 256, 0, stream>>>(Xh, Wkn, bkn, Wqn, bqn, Wvn, bvn, Kn, Qn, Vn);
  k_cmat<<<16, 256, 0, stream>>>(Kn, Qn, Wqe, Wke, CnT, CqT);
  k_dvec<<<4, 256, 0, stream>>>(Kn, Qn, bqe, bke, dnv, dqv);
  k_edge<<<1024, 256, 0, stream>>>(e, Vn, CnT, CqT, dnv, dqv, e_new, Agg);
  k_xnew<<<8, 256, 0, stream>>>(Agg, Wve, bve, x_new);
}
